// DilatedConvLayer_56547539419658
// MI455X (gfx1250) — matrix operand order repaired and checked by reading, not compiled, not run on hardware
//
#include <hip/hip_runtime.h>
#include <hip/hip_bf16.h>
#include <math.h>

// ---------------------------------------------------------------------------
// WaveNet gated dilated-conv stack for MI455X (gfx1250, wave32, WMMA).
// Layer GEMM mapped as D = W^T_eff x X^T so B-frags (activations) and D-tiles
// (outputs) are channel-contiguous per lane -> pure B128 vmem traffic.
// Activations kept in bf16 (L2-resident, 192MB), f32 accumulation via
// v_wmma_f32_16x16x32_bf16.
// ---------------------------------------------------------------------------

typedef __attribute__((ext_vector_type(16))) __bf16 v16bf;
typedef __attribute__((ext_vector_type(8)))  float  v8f;

#define T_SEQ   8192
#define BATCH   16
#define CH      64
#define NLAYER  13
#define FRAG_ELEMS_PER_LAYER (16 * 32 * 16)   // 16 frags x 32 lanes x 16 bf16

union Frag {
    v16bf v;
    uint4 q[2];
};

// --- fp32 -> bf16 convert of the input activations --------------------------
__global__ __launch_bounds__(256) void wn_convert_bf16(
    const float* __restrict__ x, __hip_bfloat16* __restrict__ o, int n4) {
  int i = blockIdx.x * 256 + threadIdx.x;
  if (i >= n4) return;
  float4 f = reinterpret_cast<const float4*>(x)[i];
  alignas(8) __hip_bfloat16 h[4];
  h[0] = __float2bfloat16(f.x); h[1] = __float2bfloat16(f.y);
  h[2] = __float2bfloat16(f.z); h[3] = __float2bfloat16(f.w);
  reinterpret_cast<uint2*>(o)[i] = *reinterpret_cast<const uint2*>(h);
}

// --- pre-swizzle conv weights into WMMA A-fragment-major bf16 layout --------
// A matrix (per layer) is W_eff^T: rows m = out-channel, cols k = 0..127
// (k<64 -> tap1 (current sample), k>=64 -> tap0 (delayed sample)).
// Fragment f = mc*4+kc; lane l; elem j uses contiguous k order (same as the B frags):
//   m = mc*16 + (l&15);  Kl = j+16*(l>>4);  k = kc*32+Kl
__global__ __launch_bounds__(256) void wn_swizzle_w(
    const float* __restrict__ Wf, const float* __restrict__ Wg,
    __hip_bfloat16* __restrict__ of, __hip_bfloat16* __restrict__ og) {
  const int per = NLAYER * FRAG_ELEMS_PER_LAYER;
  int tid = blockIdx.x * 256 + threadIdx.x;
  if (tid >= 2 * per) return;
  int s = tid / per;
  int r = tid % per;
  int j  = r & 15;
  int l  = (r >> 4) & 31;
  int f  = (r >> 9) & 15;
  int i  = r >> 13;
  int mc = f >> 2, kc = f & 3;
  int o  = mc * 16 + (l & 15);
  int kh = l >> 4;
  int Kl = j + 16 * kh;
  int k  = kc * 32 + Kl;
  int tap = (k < 64) ? 1 : 0;          // k<64: current-sample tap W[1]
  int c   = (k < 64) ? k : (k - 64);   // k>=64: delayed-sample tap W[0]
  const float* W = s ? Wg : Wf;
  float val = W[(((size_t)i * 2 + tap) * 64 + c) * 64 + o];
  (s ? og : of)[r] = __float2bfloat16(val);
}

// --- bias = h @ V (16x64 @ 64x64, trivially small) --------------------------
__global__ __launch_bounds__(256) void wn_bias(
    const float* __restrict__ h, const float* __restrict__ Vf,
    const float* __restrict__ Vg, float* __restrict__ bf_, float* __restrict__ bg_) {
  int tid = blockIdx.x * 256 + threadIdx.x;
  if (tid >= 2 * BATCH * CH) return;
  int s = tid >> 10;
  int r = tid & 1023;
  int b = r >> 6, o = r & 63;
  const float* V = s ? Vg : Vf;
  float acc = 0.f;
  #pragma unroll 8
  for (int c = 0; c < 64; ++c) acc += h[b * 64 + c] * V[c * 64 + o];
  (s ? bg_ : bf_)[r] = acc;
}

// --- one dilated causal conv layer: out[t,:] = x[t]@W1 + x[t-d]@W0 ----------
// Transposed WMMA mapping: M=out-ch, N=time, K=128. Wave holds all 16 weight
// A-fragments persistently; per 16x64 tile: 8 B128 loads, 16 WMMA, 4 B128 sts.
__global__ __launch_bounds__(256) void wn_conv(
    const __hip_bfloat16* __restrict__ xin, __hip_bfloat16* __restrict__ xout,
    const __hip_bfloat16* __restrict__ wsw, int d) {
  const int lane  = threadIdx.x & 31;
  const int nwave = gridDim.x * (blockDim.x >> 5);
  const int wave  = blockIdx.x * (blockDim.x >> 5) + (threadIdx.x >> 5);
  const int kh = lane >> 4;     // half-wave select
  const int ln = lane & 15;     // time index within tile (B/D lane = N = time)

  // Load 16 pre-swizzled weight fragments (layer A-matrix), fully coalesced.
  Frag A[16];
  const char* wb = (const char*)wsw;
  #pragma unroll
  for (int f = 0; f < 16; ++f) {
    const uint4* p = (const uint4*)(wb + f * 1024 + lane * 32);
    A[f].q[0] = p[0];
    A[f].q[1] = p[1];
  }

  const int NT = (BATCH * T_SEQ) / 16;   // 8192 tiles of 16 timesteps
  for (int tile = wave; tile < NT; tile += nwave) {
    int b  = tile >> 9;                   // 512 tiles per batch element
    int t0 = (tile & 511) << 4;
    int t  = t0 + ln;                     // this lane's timestep
    size_t row = ((size_t)b * T_SEQ + t) * CH;
    const char* xr = (const char*)xin + row * 2;

    // B fragments: kc0/1 from x[t], kc2/3 from x[t-d] (zero if t<d, causal).
    Frag Bf[4];
    {
      const uint4* p0 = (const uint4*)(xr + kh * 32);
      Bf[0].q[0] = p0[0]; Bf[0].q[1] = p0[1];
      const uint4* p1 = (const uint4*)(xr + 64 + kh * 32);
      Bf[1].q[0] = p1[0]; Bf[1].q[1] = p1[1];
    }
    if (t >= d) {
      const char* xr2 = xr - (size_t)d * (CH * 2);
      const uint4* p2 = (const uint4*)(xr2 + kh * 32);
      Bf[2].q[0] = p2[0]; Bf[2].q[1] = p2[1];
      const uint4* p3 = (const uint4*)(xr2 + 64 + kh * 32);
      Bf[3].q[0] = p3[0]; Bf[3].q[1] = p3[1];
    } else {
      uint4 z = make_uint4(0u, 0u, 0u, 0u);
      Bf[2].q[0] = z; Bf[2].q[1] = z; Bf[3].q[0] = z; Bf[3].q[1] = z;
    }

    v8f acc[4] = {};   // 4 independent accumulation chains (out-ch blocks)
    #pragma unroll
    for (int kc = 0; kc < 4; ++kc) {
      #pragma unroll
      for (int mc = 0; mc < 4; ++mc) {
        acc[mc] = __builtin_amdgcn_wmma_f32_16x16x32_bf16(
            false, A[mc * 4 + kc].v, false, Bf[kc].v,
            (short)0, acc[mc], false, false);
      }
    }

    // D tile: lane = time, elems = 8 contiguous out-channels -> B128 stores.
    #pragma unroll
    for (int mc = 0; mc < 4; ++mc) {
      alignas(16) __hip_bfloat16 hb[8];
      #pragma unroll
      for (int r2 = 0; r2 < 8; ++r2) hb[r2] = __float2bfloat16(acc[mc][r2]);
      *(uint4*)((char*)xout + row * 2 + (mc * 16 + 8 * kh) * 2) =
          *(const uint4*)hb;
    }
  }
}

// --- fused gate: out = tanh(f + biasF[b]) * sigmoid(g + biasG[b]) -----------
__global__ __launch_bounds__(256) void wn_gate(
    const __hip_bfloat16* __restrict__ fB, const __hip_bfloat16* __restrict__ gB,
    const float* __restrict__ bf_, const float* __restrict__ bg_,
    float* __restrict__ out, int n4) {
  int i = blockIdx.x * 256 + threadIdx.x;
  if (i >= n4) return;
  int base = i * 4;
  int c0  = base & 63;
  int row = base >> 6;
  int b   = row >> 13;   // row / T_SEQ
  uint2 fp = reinterpret_cast<const uint2*>(fB)[i];
  uint2 gp = reinterpret_cast<const uint2*>(gB)[i];
  const __hip_bfloat16* fh = (const __hip_bfloat16*)&fp;
  const __hip_bfloat16* gh = (const __hip_bfloat16*)&gp;
  float4 o;
  float* op = &o.x;
  #pragma unroll
  for (int e = 0; e < 4; ++e) {
    float fv = __bfloat162float(fh[e]) + bf_[b * 64 + c0 + e];
    float gv = __bfloat162float(gh[e]) + bg_[b * 64 + c0 + e];
    float s  = 1.0f / (1.0f + __expf(-gv));
    op[e] = tanhf(fv) * s;
  }
  reinterpret_cast<float4*>(out)[i] = o;
}

// ---------------------------------------------------------------------------
extern "C" void kernel_launch(void* const* d_in, const int* in_sizes, int n_in,
                              void* d_out, int out_size, void* d_ws, size_t ws_size,
                              hipStream_t stream) {
  const float* x  = (const float*)d_in[0];
  const float* h  = (const float*)d_in[1];
  const float* Wf = (const float*)d_in[2];
  const float* Wg = (const float*)d_in[3];
  const float* Vf = (const float*)d_in[4];
  const float* Vg = (const float*)d_in[5];
  float* out = (float*)d_out;

  const size_t NX = (size_t)BATCH * T_SEQ * CH;   // 8,388,608 elems
  const size_t WSZ = (size_t)NLAYER * FRAG_ELEMS_PER_LAYER;

  // Workspace carve-up (bf16 activations + swizzled weights + biases).
  __hip_bfloat16* X0  = (__hip_bfloat16*)d_ws;
  __hip_bfloat16* XfA = X0  + NX;
  __hip_bfloat16* XfB = XfA + NX;
  __hip_bfloat16* XgA = XfB + NX;
  __hip_bfloat16* XgB = XgA + NX;
  __hip_bfloat16* WfS = XgB + NX;
  __hip_bfloat16* WgS = WfS + WSZ;
  float* biasF = (float*)(WgS + WSZ);
  float* biasG = biasF + BATCH * CH;

  const int n4 = (int)(NX / 4);

  wn_convert_bf16<<<(n4 + 255) / 256, 256, 0, stream>>>(x, X0, n4);
  wn_swizzle_w<<<(int)((2 * WSZ + 255) / 256), 256, 0, stream>>>(Wf, Wg, WfS, WgS);
  wn_bias<<<8, 256, 0, stream>>>(h, Vf, Vg, biasF, biasG);

  const int convBlocks = 256;   // 2048 wave32s; 8192 tiles -> 4 tiles/wave

  // Filter stack (13 dilated causal conv layers, ping-pong in L2).
  const __hip_bfloat16* cin = X0;
  __hip_bfloat16* fbufs[2] = {XfA, XfB};
  for (int i = 0; i < NLAYER; ++i) {
    __hip_bfloat16* o = fbufs[i & 1];
    wn_conv<<<convBlocks, 256, 0, stream>>>(cin, o, WfS + (size_t)i * FRAG_ELEMS_PER_LAYER, 1 << i);
    cin = o;
  }
  const __hip_bfloat16* fFinal = cin;

  // Gate stack.
  cin = X0;
  __hip_bfloat16* gbufs[2] = {XgA, XgB};
  for (int i = 0; i < NLAYER; ++i) {
    __hip_bfloat16* o = gbufs[i & 1];
    wn_conv<<<convBlocks, 256, 0, stream>>>(cin, o, WgS + (size_t)i * FRAG_ELEMS_PER_LAYER, 1 << i);
    cin = o;
  }
  const __hip_bfloat16* gFinal = cin;

  wn_gate<<<(n4 + 255) / 256, 256, 0, stream>>>(fFinal, gFinal, biasF, biasG, out, n4);
}
